// MultiHeadAttention_8899172237976
// MI455X (gfx1250) — compile-verified, matrix operand order checked by reading, not run on hardware
//
#include <hip/hip_runtime.h>

typedef __attribute__((ext_vector_type(16))) _Float16 v16h;
typedef __attribute__((ext_vector_type(8)))  _Float16 v8h;
typedef __attribute__((ext_vector_type(8)))  float    v8f;
typedef __attribute__((ext_vector_type(4)))  unsigned int v4u;
typedef __attribute__((ext_vector_type(8)))  int      v8i_;
typedef __attribute__((ext_vector_type(4)))  int      v4i_;

#define WMMA_F16(a,b,c) \
  __builtin_amdgcn_wmma_f32_16x16x32_f16(false,(a),false,(b),(short)0,(c),false,false)

#if defined(__has_builtin)
#  if __has_builtin(__builtin_amdgcn_tensor_load_to_lds)
#    define HAVE_TDM 1
#  else
#    define HAVE_TDM 0
#  endif
#else
#  define HAVE_TDM 0
#endif

static constexpr int S_LEN = 2048;
static constexpr int EMB   = 1024;
static constexpr int NH    = 16;
static constexpr int DH    = 64;
static constexpr int BATCH = 2;
static constexpr int MTOT  = BATCH * S_LEN;   // 4096 rows total
static constexpr int BH    = BATCH * NH;      // 32 (batch, head) pairs
static constexpr int KCH   = 256;             // K elems per staged weight chunk
static constexpr int NKCH  = EMB / KCH;       // 4 chunks

// Load a 16-bit A/B-style WMMA fragment (16x32 MxK for A, or B^T rows for B).
// p must already point at (row m, element kbase) where kbase = (lane<16)?0:8.
// Per ISA layout: elements 0..7 = K kb..kb+7, elements 8..15 = K kb+16..kb+23.
__device__ __forceinline__ v16h frag_ld(const _Float16* p) {
  v8h lo = *(const v8h*)(p);
  v8h hi = *(const v8h*)(p + 16);
  v16h a;
#pragma unroll
  for (int i = 0; i < 8; ++i) { a[i] = lo[i]; a[i + 8] = hi[i]; }
  return a;
}

#if HAVE_TDM
// Issue a TDM 2D-tile load: tile_rows rows of tile_elems_x f16 elements each,
// rows separated by row_stride elements in global memory, packed contiguously
// into LDS at lds_off. D# layout per CDNA5 ISA.
__device__ __forceinline__ void tdm_load_2d(unsigned lds_off,
                                            const _Float16* gsrc,
                                            unsigned tile_elems_x,
                                            unsigned tile_rows,
                                            unsigned row_stride) {
  unsigned long long ga = (unsigned long long)gsrc;
  v4u g0;
  g0[0] = 1u;                                        // count=1, user descriptor
  g0[1] = lds_off;                                   // lds_addr (bytes)
  g0[2] = (unsigned)ga;                              // global_addr[31:0]
  g0[3] = (unsigned)((ga >> 32) & 0x01FFFFFFu)       // global_addr[56:32]
        | (2u << 30);                                // type = 2 ("image")
  v8i_ g1;
  g1[0] = (int)(1u << 16);                           // data_size=1 -> 2 bytes
  g1[1] = (int)((tile_elems_x & 0xFFFFu) << 16);     // tensor_dim0[15:0] @63:48
  g1[2] = (int)(((tile_elems_x >> 16) & 0xFFFFu)     // tensor_dim0[31:16]
        |       ((tile_rows & 0xFFFFu) << 16));      // tensor_dim1[15:0]
  g1[3] = (int)(((tile_rows >> 16) & 0xFFFFu)        // tensor_dim1[31:16]
        |       ((tile_elems_x & 0xFFFFu) << 16));   // tile_dim0
  g1[4] = (int)(tile_rows & 0xFFFFu);                // tile_dim1 (tile_dim2=0)
  g1[5] = (int)row_stride;                           // tensor_dim0_stride[31:0]
  g1[6] = 0;                                         // stride hi / dim1_stride
  g1[7] = 0;
  v4i_ z4 = (v4i_)0;
#if __clang_major__ >= 23
  v8i_ z8 = (v8i_)0;
  __builtin_amdgcn_tensor_load_to_lds(g0, g1, z4, z4, z8, 0);
#else
  __builtin_amdgcn_tensor_load_to_lds(g0, g1, z4, z4, 0);
#endif
}
#endif

// ---------------------------------------------------------------------------
// Conversions
// ---------------------------------------------------------------------------
__global__ void cvt_f32_to_f16(const float* __restrict__ in,
                               _Float16* __restrict__ out, int n) {
  int i = blockIdx.x * blockDim.x + threadIdx.x;
  if (i < n) out[i] = (_Float16)in[i];
}

// Transpose-convert the four 1024x1024 weights: out[n][k] = (f16) in[k][n]
__global__ void cvt_w_transpose(const float* __restrict__ w0,
                                const float* __restrict__ w1,
                                const float* __restrict__ w2,
                                const float* __restrict__ w3,
                                _Float16* __restrict__ o0,
                                _Float16* __restrict__ o1,
                                _Float16* __restrict__ o2,
                                _Float16* __restrict__ o3) {
  const float* in  = (blockIdx.y == 0) ? w0 : (blockIdx.y == 1) ? w1
                   : (blockIdx.y == 2) ? w2 : w3;
  _Float16*   out  = (blockIdx.y == 0) ? o0 : (blockIdx.y == 1) ? o1
                   : (blockIdx.y == 2) ? o2 : o3;
  int i = blockIdx.x * 256 + threadIdx.x;      // output index: n*1024 + k
  int n = i >> 10;
  int k = i & 1023;
  out[i] = (_Float16)in[(size_t)k * EMB + n];
}

// ---------------------------------------------------------------------------
// Weight-panel staging: TDM double-buffered (preferred) or cooperative copy.
// Panel = 64 weight rows x KCH k-elements, packed [64][KCH] in LDS.
// ---------------------------------------------------------------------------
__device__ __forceinline__ void stage_issue(_Float16* buf, const _Float16* Wt,
                                            int nBase, int chunk) {
#if HAVE_TDM
  tdm_load_2d((unsigned)(unsigned long long)buf,
              Wt + (size_t)nBase * EMB + chunk * KCH, KCH, 64, EMB);
#else
  (void)buf; (void)Wt; (void)nBase; (void)chunk;
#endif
}

// ---------------------------------------------------------------------------
// Fused QKV projection: z=0 -> Q[bh][s][d], z=1 -> K[bh][s][d], z=2 -> Vt[bh][d][s]
// X: [4096][1024] f16 row-major, W*t: [N=1024][K=1024] f16 (transposed weights)
// ---------------------------------------------------------------------------
__global__ void gemm_qkv(const _Float16* __restrict__ X,
                         const _Float16* __restrict__ Wqt,
                         const _Float16* __restrict__ Wkt,
                         const _Float16* __restrict__ Wvt,
                         _Float16* __restrict__ Qh,
                         _Float16* __restrict__ Kh,
                         _Float16* __restrict__ Vt) {
  __shared__ _Float16 ldsB[2][64 * KCH];      // 2 x 32 KiB double buffer

  const int lane  = threadIdx.x & 31;
  const int wave  = threadIdx.x >> 5;
  const int mBase = blockIdx.x * 128 + wave * 16;
  const int nBase = blockIdx.y * 64;
  const int z     = blockIdx.z;
  const _Float16* Wt = (z == 0) ? Wqt : (z == 1) ? Wkt : Wvt;

  const int mrow = lane & 15;
  const int kb   = (lane < 16) ? 0 : 8;
  const int hi   = lane >> 4;
  const int nCol = lane & 15;

  v8f acc[4];
#pragma unroll
  for (int t = 0; t < 4; ++t) acc[t] = (v8f){0,0,0,0,0,0,0,0};

  const _Float16* arow = X + (size_t)(mBase + mrow) * EMB + kb;

#if HAVE_TDM
  if (wave == 0) stage_issue(ldsB[0], Wt, nBase, 0);   // prologue chunk 0
#endif
  for (int c = 0; c < NKCH; ++c) {
#if HAVE_TDM
    if (wave == 0) {
      if (c + 1 < NKCH) {
        stage_issue(ldsB[(c + 1) & 1], Wt, nBase, c + 1);
        __builtin_amdgcn_s_wait_tensorcnt(1);          // chunk c ready
      } else {
        __builtin_amdgcn_s_wait_tensorcnt(0);
      }
    }
#else
    // cooperative synchronous copy of chunk c (fallback, no TDM builtin)
    for (int i = threadIdx.x; i < 64 * KCH / 8; i += 256) {
      int rr = i / (KCH / 8), cc = (i % (KCH / 8)) * 8;
      *(v8h*)&ldsB[c & 1][rr * KCH + cc] =
          *(const v8h*)(Wt + (size_t)(nBase + rr) * EMB + c * KCH + cc);
    }
#endif
    __syncthreads();                                   // chunk c visible
    const _Float16* bbase = &ldsB[c & 1][0];
    for (int kk = 0; kk < KCH; kk += 32) {
      v16h a = frag_ld(arow + c * KCH + kk);
#pragma unroll
      for (int t = 0; t < 4; ++t) {
        v16h b = frag_ld(bbase + (t * 16 + mrow) * KCH + kk + kb);
        acc[t] = WMMA_F16(a, b, acc[t]);
      }
    }
    __syncthreads();                                   // done reading buf[c&1]
  }

#pragma unroll
  for (int t = 0; t < 4; ++t) {
#pragma unroll
    for (int r = 0; r < 8; ++r) {
      int row = mBase + hi * 8 + r;           // [0,4096)
      int col = nBase + t * 16 + nCol;        // [0,1024)
      int b   = row >> 11, s = row & (S_LEN - 1);
      int h   = col >> 6,  d = col & (DH - 1);
      int bh  = b * NH + h;
      _Float16 val = (_Float16)acc[t][r];
      if (z == 0)      Qh[((size_t)bh * S_LEN + s) * DH + d] = val;
      else if (z == 1) Kh[((size_t)bh * S_LEN + s) * DH + d] = val;
      else             Vt[((size_t)bh * DH + d) * S_LEN + s] = val;
    }
  }
}

// ---------------------------------------------------------------------------
// Flash attention: one wave per 16-query tile, online softmax over 32-key chunks
// ---------------------------------------------------------------------------
__global__ void attn_flash(const _Float16* __restrict__ Qh,
                           const _Float16* __restrict__ Kh,
                           const _Float16* __restrict__ Vt,
                           _Float16* __restrict__ Oh) {
  __shared__ _Float16 lds_p[8][16][32];       // per-wave P staging tile

  const int lane = threadIdx.x & 31;
  const int wave = threadIdx.x >> 5;
  const int bh   = blockIdx.y;
  const int q0   = blockIdx.x * 128 + wave * 16;

  const int mrow = lane & 15;
  const int kb   = (lane < 16) ? 0 : 8;
  const int hi   = lane >> 4;
  const int nCol = lane & 15;

  const _Float16* Qb = Qh + (size_t)bh * S_LEN * DH;
  const _Float16* Kb = Kh + (size_t)bh * S_LEN * DH;
  const _Float16* Vb = Vt + (size_t)bh * DH * S_LEN;

  // Q fragments for this wave's 16 rows, Dh split 0..31 / 32..63
  v16h qf0 = frag_ld(Qb + (size_t)(q0 + mrow) * DH + kb);
  v16h qf1 = frag_ld(Qb + (size_t)(q0 + mrow) * DH + 32 + kb);

  v8f acc[4];
#pragma unroll
  for (int t = 0; t < 4; ++t) acc[t] = (v8f){0,0,0,0,0,0,0,0};
  float mrun[8], lrun[8];
#pragma unroll
  for (int r = 0; r < 8; ++r) { mrun[r] = -1e30f; lrun[r] = 0.0f; }

  const int cmax = (q0 + 15) >> 5;            // last 32-key chunk touching diag
  for (int c = 0; c <= cmax; ++c) {
    const int k0 = c * 32;

    // ---- scores: two 16x16 tiles covering keys k0..k0+31 ----
    v8f sc[2];
#pragma unroll
    for (int t = 0; t < 2; ++t) {
      const _Float16* krow = Kb + (size_t)(k0 + t * 16 + mrow) * DH + kb;
      v16h b0 = frag_ld(krow);
      v16h b1 = frag_ld(krow + 32);
      v8f s = (v8f){0,0,0,0,0,0,0,0};
      s = WMMA_F16(qf0, b0, s);
      s = WMMA_F16(qf1, b1, s);
#pragma unroll
      for (int r = 0; r < 8; ++r) {
        float v  = s[r] * 0.125f;             // 1/sqrt(64)
        int   qg = q0 + hi * 8 + r;
        int   kg = k0 + t * 16 + nCol;
        s[r] = (kg > qg) ? -1e30f : v;        // causal mask
      }
      sc[t] = s;
    }

    // ---- online softmax update + stage P into LDS ----
#pragma unroll
    for (int r = 0; r < 8; ++r) {
      float mx = fmaxf(sc[0][r], sc[1][r]);
      mx = fmaxf(mx, __shfl_xor(mx, 1));
      mx = fmaxf(mx, __shfl_xor(mx, 2));
      mx = fmaxf(mx, __shfl_xor(mx, 4));
      mx = fmaxf(mx, __shfl_xor(mx, 8));
      float mnew  = fmaxf(mrun[r], mx);
      float alpha = __expf(mrun[r] - mnew);
      float p0 = __expf(sc[0][r] - mnew);
      float p1 = __expf(sc[1][r] - mnew);
      float rs = p0 + p1;
      rs += __shfl_xor(rs, 1);
      rs += __shfl_xor(rs, 2);
      rs += __shfl_xor(rs, 4);
      rs += __shfl_xor(rs, 8);
      lrun[r] = lrun[r] * alpha + rs;
      mrun[r] = mnew;
#pragma unroll
      for (int t = 0; t < 4; ++t) acc[t][r] *= alpha;
      lds_p[wave][hi * 8 + r][nCol]      = (_Float16)p0;
      lds_p[wave][hi * 8 + r][16 + nCol] = (_Float16)p1;
    }
    // wave-private LDS tile; DS ops are in-order per wave, just fence compiler
    asm volatile("s_wait_dscnt 0" ::: "memory");
    __builtin_amdgcn_wave_barrier();

    // ---- reload P in A-fragment layout and accumulate P @ V ----
    v16h pf = frag_ld(&lds_p[wave][mrow][kb]);
#pragma unroll
    for (int t = 0; t < 4; ++t) {
      const _Float16* vrow = Vb + (size_t)(t * 16 + mrow) * S_LEN + k0 + kb;
      v16h bf = frag_ld(vrow);
      acc[t] = WMMA_F16(pf, bf, acc[t]);
    }
    asm volatile("" ::: "memory");            // order LDS reads vs next writes
  }

  // ---- finalize and store merged-head output [4096][1024] f16 ----
  const int b = bh >> 4, h = bh & 15;
#pragma unroll
  for (int t = 0; t < 4; ++t) {
#pragma unroll
    for (int r = 0; r < 8; ++r) {
      float o = acc[t][r] / lrun[r];
      int s = q0 + hi * 8 + r;
      int d = t * 16 + nCol;
      Oh[((size_t)(b * S_LEN + s)) * EMB + h * DH + d] = (_Float16)o;
    }
  }
}

// ---------------------------------------------------------------------------
// Output projection: Y = Oh @ Wp + bp, f32 output (TDM-staged weight panel)
// ---------------------------------------------------------------------------
__global__ void gemm_out(const _Float16* __restrict__ Oh,
                         const _Float16* __restrict__ Wpt,
                         const float* __restrict__ bp,
                         float* __restrict__ Y) {
  __shared__ _Float16 ldsB[2][64 * KCH];

  const int lane  = threadIdx.x & 31;
  const int wave  = threadIdx.x >> 5;
  const int mBase = blockIdx.x * 128 + wave * 16;
  const int nBase = blockIdx.y * 64;

  const int mrow = lane & 15;
  const int kb   = (lane < 16) ? 0 : 8;
  const int hi   = lane >> 4;
  const int nCol = lane & 15;

  v8f acc[4];
#pragma unroll
  for (int t = 0; t < 4; ++t) acc[t] = (v8f){0,0,0,0,0,0,0,0};

  const _Float16* arow = Oh + (size_t)(mBase + mrow) * EMB + kb;

#if HAVE_TDM
  if (wave == 0) stage_issue(ldsB[0], Wpt, nBase, 0);
#endif
  for (int c = 0; c < NKCH; ++c) {
#if HAVE_TDM
    if (wave == 0) {
      if (c + 1 < NKCH) {
        stage_issue(ldsB[(c + 1) & 1], Wpt, nBase, c + 1);
        __builtin_amdgcn_s_wait_tensorcnt(1);
      } else {
        __builtin_amdgcn_s_wait_tensorcnt(0);
      }
    }
#else
    for (int i = threadIdx.x; i < 64 * KCH / 8; i += 256) {
      int rr = i / (KCH / 8), cc = (i % (KCH / 8)) * 8;
      *(v8h*)&ldsB[c & 1][rr * KCH + cc] =
          *(const v8h*)(Wpt + (size_t)(nBase + rr) * EMB + c * KCH + cc);
    }
#endif
    __syncthreads();
    const _Float16* bbase = &ldsB[c & 1][0];
    for (int kk = 0; kk < KCH; kk += 32) {
      v16h a = frag_ld(arow + c * KCH + kk);
#pragma unroll
      for (int t = 0; t < 4; ++t) {
        v16h b = frag_ld(bbase + (t * 16 + mrow) * KCH + kk + kb);
        acc[t] = WMMA_F16(a, b, acc[t]);
      }
    }
    __syncthreads();
  }

#pragma unroll
  for (int t = 0; t < 4; ++t) {
    float bias = bp[nBase + t * 16 + nCol];
#pragma unroll
    for (int r = 0; r < 8; ++r) {
      int row = mBase + hi * 8 + r;
      int col = nBase + t * 16 + nCol;
      Y[(size_t)row * EMB + col] = acc[t][r] + bias;
    }
  }
}

// ---------------------------------------------------------------------------
extern "C" void kernel_launch(void* const* d_in, const int* in_sizes, int n_in,
                              void* d_out, int out_size, void* d_ws, size_t ws_size,
                              hipStream_t stream) {
  (void)in_sizes; (void)n_in; (void)out_size; (void)ws_size;
  // setup_inputs order: x, Wk, Wq, Wv, Wp, bp
  const float* x   = (const float*)d_in[0];
  const float* Wk  = (const float*)d_in[1];
  const float* Wq  = (const float*)d_in[2];
  const float* Wv  = (const float*)d_in[3];
  const float* Wp  = (const float*)d_in[4];
  const float* bp  = (const float*)d_in[5];
  float* out = (float*)d_out;

  char* ws = (char*)d_ws;
  size_t off = 0;
  auto carve = [&](size_t bytes) -> void* {
    void* p = ws + off;
    off += (bytes + 255) & ~(size_t)255;
    return p;
  };
  _Float16* xh  = (_Float16*)carve((size_t)MTOT * EMB * 2);   // 8 MiB
  _Float16* Wqt = (_Float16*)carve((size_t)EMB * EMB * 2);    // 2 MiB
  _Float16* Wkt = (_Float16*)carve((size_t)EMB * EMB * 2);
  _Float16* Wvt = (_Float16*)carve((size_t)EMB * EMB * 2);
  _Float16* Wpt = (_Float16*)carve((size_t)EMB * EMB * 2);
  _Float16* Qh  = (_Float16*)carve((size_t)BH * S_LEN * DH * 2);  // 8 MiB
  _Float16* Kh  = (_Float16*)carve((size_t)BH * S_LEN * DH * 2);
  _Float16* Vt  = (_Float16*)carve((size_t)BH * DH * S_LEN * 2);
  _Float16* Oh  = (_Float16*)carve((size_t)MTOT * EMB * 2);       // 8 MiB

  const int nX = MTOT * EMB;                  // 4,194,304
  cvt_f32_to_f16<<<(nX + 255) / 256, 256, 0, stream>>>(x, xh, nX);
  cvt_w_transpose<<<dim3((EMB * EMB) / 256, 4), 256, 0, stream>>>(
      Wq, Wk, Wv, Wp, Wqt, Wkt, Wvt, Wpt);

  // QKV projections: M=4096 rows (32 blocks x 128), N=1024 (16 blocks x 64)
  gemm_qkv<<<dim3(MTOT / 128, EMB / 64, 3), 256, 0, stream>>>(
      xh, Wqt, Wkt, Wvt, Qh, Kh, Vt);

  // Attention: 8 waves x 16 queries per block -> 16 blocks over S, 32 (b,h)
  attn_flash<<<dim3(S_LEN / 128, BH), 256, 0, stream>>>(Qh, Kh, Vt, Oh);

  // Output projection + bias
  gemm_out<<<dim3(MTOT / 128, EMB / 64), 256, 0, stream>>>(Oh, Wpt, bp, out);
}
